// GraphTGI_classifier_34385508171922
// MI455X (gfx1250) — compile-verified, matrix operand order checked by reading, not run on hardware
//
#include <hip/hip_runtime.h>

#define N_TF 50000
#define N_TG 200000
#define NE   1000000
#define SLOPE 0.2f

typedef __attribute__((ext_vector_type(16))) __bf16 v16bf;
typedef __attribute__((ext_vector_type(8)))  __bf16 v8bf;
typedef __attribute__((ext_vector_type(8)))  float  v8f;

__device__ __forceinline__ unsigned short f2bf_bits(float f) {
  // round-to-nearest-even f32 -> bf16 (bulk passes only; not in GEMM loop)
  unsigned u = __builtin_bit_cast(unsigned, f);
  return (unsigned short)((u + 0x7FFFu + ((u >> 16) & 1u)) >> 16);
}
__device__ __forceinline__ __bf16 f2bf(float f) {
  unsigned short s = f2bf_bits(f);
  return __builtin_bit_cast(__bf16, s);
}
__device__ __forceinline__ float leaky(float x) { return x >= 0.f ? x : SLOPE * x; }
__device__ __forceinline__ void atomAddF(float* p, float v) {
  __hip_atomic_fetch_add(p, v, __ATOMIC_RELAXED, __HIP_MEMORY_SCOPE_AGENT);
}

__global__ void zero_f32(float* __restrict__ p, int n) {
  int i = blockIdx.x * blockDim.x + threadIdx.x;
  if (i < n) p[i] = 0.f;
}

__global__ void deg_count(const int* __restrict__ idx, float* __restrict__ deg, int n) {
  int i = blockIdx.x * blockDim.x + threadIdx.x;
  if (i < n) atomAddF(&deg[idx[i]], 1.0f);
}

__global__ void deg_rsqrt(float* __restrict__ deg, int n) {
  int i = blockIdx.x * blockDim.x + threadIdx.x;
  if (i < n) deg[i] = rsqrtf(fmaxf(deg[i], 1.0f));
}

// Wt[n*K + k] = bf16(W[k*N + n])  (transpose: B-operand k runs contiguous)
__global__ void wt_conv(const float* __restrict__ W, __bf16* __restrict__ Wt, int K, int N) {
  int i = blockIdx.x * blockDim.x + threadIdx.x;
  if (i >= K * N) return;
  int n = i / K, k = i - n * K;
  Wt[i] = f2bf(W[(size_t)k * N + n]);
}

// bulk f32 -> bf16 (4 elements / thread, packed 8-byte stores)
__global__ void feat2bf4(const float* __restrict__ x, __bf16* __restrict__ xb, int n4) {
  int i = blockIdx.x * blockDim.x + threadIdx.x;
  if (i >= n4) return;
  float4 v = ((const float4*)x)[i];
  unsigned lo = (unsigned)f2bf_bits(v.x) | ((unsigned)f2bf_bits(v.y) << 16);
  unsigned hi = (unsigned)f2bf_bits(v.z) | ((unsigned)f2bf_bits(v.w) << 16);
  ((uint2*)xb)[i] = make_uint2(lo, hi);
}

// C[M, NT*16] = diag(rsc) * (A_bf16[M,K] @ Wt^T), f32 accumulate via WMMA.
// One wave = one 16-row panel across all NT column tiles (A frag reused NT times).
// Wt staged in LDS with row padding (K+8 elems) to spread ds banks.
template <int NT>
__global__ __launch_bounds__(256) void gemm_wmma_bf(
    const __bf16* __restrict__ A, const float* __restrict__ rsc,
    const __bf16* __restrict__ Wt, float* __restrict__ C,
    int M, int K) {
  constexpr int N = NT * 16;
  extern __shared__ __bf16 wlds[];          // [N][K+8]
  const int KP = K + 8;
  const int tid = threadIdx.x;

  // cooperative stage: global Wt[n*K+k] -> LDS wlds[n*KP+k]
  for (int i = tid; i < N * K; i += 256) {
    int n = i / K, k = i - n * K;
    wlds[n * KP + k] = Wt[i];
  }
  __syncthreads();

  const int lane = tid & 31;
  const int wave = tid >> 5;
  const int tilesM = M >> 4;                 // M % 16 == 0 at all call sites
  const int tm = blockIdx.x * 8 + wave;      // wave-uniform
  if (tm < tilesM) {                         // uniform guard: EXEC stays all-1s
    const int hid = lane >> 4;               // wave half selects K sub-ranges
    const int l15 = lane & 15;
    const __bf16* arow = A + (size_t)((tm << 4) + l15) * K;
    const __bf16* brow = wlds + (size_t)l15 * KP;
    v8f acc[NT];
#pragma unroll
    for (int t = 0; t < NT; ++t) acc[t] = v8f{};

    for (int k0 = 0; k0 < K; k0 += 32) {
      const int ka = k0 + (hid << 3);        // 16-bit A/B layout: [ka,ka+8)
      const int kb = ka + 16;                //                    [kb,kb+8)
      v8bf a0 = *(const v8bf*)(arow + ka);   // global_load_b128
      v8bf a1 = *(const v8bf*)(arow + kb);
      v16bf a;
#pragma unroll
      for (int i = 0; i < 8; ++i) { a[i] = a0[i]; a[i + 8] = a1[i]; }
#pragma unroll
      for (int t = 0; t < NT; ++t) {
        const __bf16* bp = brow + (size_t)t * 16 * KP;
        v8bf b0 = *(const v8bf*)(bp + ka);   // ds_load_b128
        v8bf b1 = *(const v8bf*)(bp + kb);
        v16bf b;
#pragma unroll
        for (int i = 0; i < 8; ++i) { b[i] = b0[i]; b[i + 8] = b1[i]; }
        acc[t] = __builtin_amdgcn_wmma_f32_16x16x32_bf16(
            false, a, false, b, (short)0, acc[t], false, false);
      }
    }

    // epilogue: row scale by rsqrt(deg_o); C/D layout row = mrow + r, col = l15
    const int mrow = (tm << 4) + (hid << 3);
    float sc[8];
#pragma unroll
    for (int r = 0; r < 8; ++r) sc[r] = rsc[mrow + r];
#pragma unroll
    for (int t = 0; t < NT; ++t) {
      float* crow = C + (size_t)mrow * N + t * 16 + l15;
#pragma unroll
      for (int r = 0; r < 8; ++r) crow[(size_t)r * N] = acc[t][r] * sc[r];
    }
  }
}

// agg[dst[e]] += h[src[e]] ; one thread per (edge, 4-feature chunk)
__global__ void scatter_add(const float* __restrict__ h, const int* __restrict__ src,
                            const int* __restrict__ dst, float* __restrict__ agg,
                            int nedge, int fo) {
  const int chunks = fo >> 2;
  int gid = blockIdx.x * blockDim.x + threadIdx.x;
  int e = gid / chunks;
  if (e >= nedge) return;
  int c = (gid - e * chunks) << 2;
  int sidx = src[e], didx = dst[e];
  const float4 v = *(const float4*)(h + (size_t)sidx * fo + c);
  float* p = agg + (size_t)didx * fo + c;
  atomAddF(p + 0, v.x);
  atomAddF(p + 1, v.y);
  atomAddF(p + 2, v.z);
  atomAddF(p + 3, v.w);
}

// out = maybe_leaky( leaky(agg1*s1+b1) + leaky(agg2*s2+b2) ) ; elementwise, in-place safe
__global__ void finalize2(const float* __restrict__ agg1, const float* __restrict__ s1,
                          const float* __restrict__ b1,
                          const float* __restrict__ agg2, const float* __restrict__ s2,
                          const float* __restrict__ b2,
                          float* __restrict__ out, int M, int fo, int extra) {
  int i = blockIdx.x * blockDim.x + threadIdx.x;
  if (i >= M * fo) return;
  int m = i / fo, f = i - m * fo;
  float v = leaky(agg1[i] * s1[m] + b1[f]) + leaky(agg2[i] * s2[m] + b2[f]);
  out[i] = extra ? leaky(v) : v;
}

// sigmoid(concat(h_tf[src], h_tg[dst]) @ Wc + bc) over [act; rep] edges
__global__ void decoder(const float* __restrict__ htf, const float* __restrict__ htg,
                        const int* __restrict__ src_a, const int* __restrict__ dst_a,
                        const int* __restrict__ src_r, const int* __restrict__ dst_r,
                        const float* __restrict__ Wc, const float* __restrict__ bc,
                        float* __restrict__ out, int nedge) {
  int e = blockIdx.x * blockDim.x + threadIdx.x;
  if (e >= 2 * nedge) return;
  int ee = (e < nedge) ? e : e - nedge;
  int si = (e < nedge) ? src_a[ee] : src_r[ee];
  int di = (e < nedge) ? dst_a[ee] : dst_r[ee];
  const float4* a4 = (const float4*)(htf + (size_t)si * 32);
  const float4* b4 = (const float4*)(htg + (size_t)di * 32);
  float acc0 = bc[0], acc1 = bc[1];
#pragma unroll
  for (int q = 0; q < 8; ++q) {
    float4 v = a4[q];
    const float* wq = Wc + 8 * q;
    acc0 += v.x * wq[0] + v.y * wq[2] + v.z * wq[4] + v.w * wq[6];
    acc1 += v.x * wq[1] + v.y * wq[3] + v.z * wq[5] + v.w * wq[7];
  }
#pragma unroll
  for (int q = 0; q < 8; ++q) {
    float4 v = b4[q];
    const float* wq = Wc + 64 + 8 * q;
    acc0 += v.x * wq[0] + v.y * wq[2] + v.z * wq[4] + v.w * wq[6];
    acc1 += v.x * wq[1] + v.y * wq[3] + v.z * wq[5] + v.w * wq[7];
  }
  out[2 * e + 0] = 1.f / (1.f + __expf(-acc0));
  out[2 * e + 1] = 1.f / (1.f + __expf(-acc1));
}

extern "C" void kernel_launch(void* const* d_in, const int* in_sizes, int n_in,
                              void* d_out, int out_size, void* d_ws, size_t ws_size,
                              hipStream_t stream) {
  (void)in_sizes; (void)n_in; (void)out_size; (void)ws_size;
  const int SIZES[4] = {256, 128, 64, 32};

  const float* x_tf = (const float*)d_in[0];
  const float* x_tg = (const float*)d_in[1];
  auto Wp = [&](int l, int et) { return (const float*)d_in[2 + l * 8 + et * 2]; };
  auto Bp = [&](int l, int et) { return (const float*)d_in[2 + l * 8 + et * 2 + 1]; };
  const float* Wc = (const float*)d_in[26];
  const float* bc = (const float*)d_in[27];
  const int* esrc[4] = {(const int*)d_in[28], (const int*)d_in[30],
                        (const int*)d_in[32], (const int*)d_in[34]};
  const int* edst[4] = {(const int*)d_in[29], (const int*)d_in[31],
                        (const int*)d_in[33], (const int*)d_in[35]};

  // ---- workspace carve (~490 MB) ----
  char* w = (char*)d_ws;
  auto carve = [&](size_t bytes) -> void* {
    void* p = (void*)w; w += (bytes + 255) & ~(size_t)255; return p;
  };
  const int nsrc[4] = {N_TF, N_TF, N_TG, N_TG};
  const int ndst[4] = {N_TG, N_TG, N_TF, N_TF};
  float* rsc_src[4]; float* rsc_dst[4];
  for (int r = 0; r < 4; ++r) {
    rsc_src[r] = (float*)carve((size_t)nsrc[r] * 4);
    rsc_dst[r] = (float*)carve((size_t)ndst[r] * 4);
  }
  __bf16* Wt[3][4];
  for (int l = 0; l < 3; ++l)
    for (int et = 0; et < 4; ++et)
      Wt[l][et] = (__bf16*)carve((size_t)SIZES[l] * SIZES[l + 1] * 2);
  float* tg[2]; float* tf[2];
  for (int i = 0; i < 2; ++i) tg[i] = (float*)carve((size_t)N_TG * 128 * 4);
  for (int i = 0; i < 2; ++i) tf[i] = (float*)carve((size_t)N_TF * 128 * 4);
  float*  hbuf  = (float*)carve((size_t)N_TG * 128 * 4);
  __bf16* xb_tf = (__bf16*)carve((size_t)N_TF * 256 * 2);
  __bf16* xb_tg = (__bf16*)carve((size_t)N_TG * 256 * 2);

  const int B = 256;
  auto g1 = [&](long long n) { return dim3((unsigned)((n + B - 1) / B)); };

  // ---- degrees (layer-invariant) ----
  for (int r = 0; r < 4; ++r) {
    zero_f32<<<g1(nsrc[r]), B, 0, stream>>>(rsc_src[r], nsrc[r]);
    zero_f32<<<g1(ndst[r]), B, 0, stream>>>(rsc_dst[r], ndst[r]);
    deg_count<<<g1(NE), B, 0, stream>>>(esrc[r], rsc_src[r], NE);
    deg_count<<<g1(NE), B, 0, stream>>>(edst[r], rsc_dst[r], NE);
    deg_rsqrt<<<g1(nsrc[r]), B, 0, stream>>>(rsc_src[r], nsrc[r]);
    deg_rsqrt<<<g1(ndst[r]), B, 0, stream>>>(rsc_dst[r], ndst[r]);
  }
  // ---- weights -> transposed bf16 ----
  for (int l = 0; l < 3; ++l)
    for (int et = 0; et < 4; ++et) {
      int kn = SIZES[l] * SIZES[l + 1];
      wt_conv<<<g1(kn), B, 0, stream>>>(Wp(l, et), Wt[l][et], SIZES[l], SIZES[l + 1]);
    }

  auto launch_gemm = [&](const __bf16* A, const float* rsc, const __bf16* Wtp,
                         float* C, int M, int fi, int fo) {
    dim3 grid((unsigned)((M / 16 + 7) / 8));
    size_t lds = (size_t)fo * (fi + 8) * sizeof(__bf16);  // <= 67.6 KB, OK on gfx1250
    switch (fo) {
      case 128: gemm_wmma_bf<8><<<grid, 256, lds, stream>>>(A, rsc, Wtp, C, M, fi); break;
      case 64:  gemm_wmma_bf<4><<<grid, 256, lds, stream>>>(A, rsc, Wtp, C, M, fi); break;
      default:  gemm_wmma_bf<2><<<grid, 256, lds, stream>>>(A, rsc, Wtp, C, M, fi); break;
    }
  };

  const float* cur_tf = x_tf;
  const float* cur_tg = x_tg;
  for (int l = 0; l < 3; ++l) {
    const int fi = SIZES[l], fo = SIZES[l + 1];
    const int extra = (l < 2) ? 1 : 0;
    // features -> bf16 once per layer per node type (unscaled; row scale commutes
    // with X@W and is applied in the GEMM epilogue per relation)
    feat2bf4<<<g1((long long)N_TF * fi / 4), B, 0, stream>>>(cur_tf, xb_tf, N_TF * fi / 4);
    feat2bf4<<<g1((long long)N_TG * fi / 4), B, 0, stream>>>(cur_tg, xb_tg, N_TG * fi / 4);

    float* aggA = tg[l & 1];      // raw f32 feature buffers are free after feat2bf
    float* aggB = tg[(l & 1) ^ 1];
    float* aggC = tf[l & 1];
    float* aggD = tf[(l & 1) ^ 1];

    // ---- dst = tg: relations act(0), rep(1); src features = xb_tf ----
    launch_gemm(xb_tf, rsc_src[0], Wt[l][0], hbuf, N_TF, fi, fo);
    zero_f32<<<g1((long long)N_TG * fo), B, 0, stream>>>(aggA, N_TG * fo);
    scatter_add<<<g1((long long)NE * (fo / 4)), B, 0, stream>>>(hbuf, esrc[0], edst[0], aggA, NE, fo);
    launch_gemm(xb_tf, rsc_src[1], Wt[l][1], hbuf, N_TF, fi, fo);
    zero_f32<<<g1((long long)N_TG * fo), B, 0, stream>>>(aggB, N_TG * fo);
    scatter_add<<<g1((long long)NE * (fo / 4)), B, 0, stream>>>(hbuf, esrc[1], edst[1], aggB, NE, fo);
    finalize2<<<g1((long long)N_TG * fo), B, 0, stream>>>(aggA, rsc_dst[0], Bp(l, 0),
                                                          aggB, rsc_dst[1], Bp(l, 1),
                                                          aggA, N_TG, fo, extra);
    // ---- dst = tf: relations afb(2), rfb(3); src features = xb_tg ----
    launch_gemm(xb_tg, rsc_src[2], Wt[l][2], hbuf, N_TG, fi, fo);
    zero_f32<<<g1((long long)N_TF * fo), B, 0, stream>>>(aggC, N_TF * fo);
    scatter_add<<<g1((long long)NE * (fo / 4)), B, 0, stream>>>(hbuf, esrc[2], edst[2], aggC, NE, fo);
    launch_gemm(xb_tg, rsc_src[3], Wt[l][3], hbuf, N_TG, fi, fo);
    zero_f32<<<g1((long long)N_TF * fo), B, 0, stream>>>(aggD, N_TF * fo);
    scatter_add<<<g1((long long)NE * (fo / 4)), B, 0, stream>>>(hbuf, esrc[3], edst[3], aggD, NE, fo);
    finalize2<<<g1((long long)N_TF * fo), B, 0, stream>>>(aggC, rsc_dst[2], Bp(l, 2),
                                                          aggD, rsc_dst[3], Bp(l, 3),
                                                          aggC, N_TF, fo, extra);
    cur_tg = aggA;
    cur_tf = aggC;
  }

  // ---- decoder over [act; rep] edges ----
  decoder<<<g1((long long)2 * NE), B, 0, stream>>>(cur_tf, cur_tg,
                                                   esrc[0], edst[0], esrc[1], edst[1],
                                                   Wc, bc, (float*)d_out, NE);
}
